// MMDLoss_87875030876301
// MI455X (gfx1250) — matrix operand order repaired and checked by reading, not compiled, not run on hardware
//
#include <hip/hip_runtime.h>
#include <hip/hip_bf16.h>

// ---------------------------------------------------------------------------
// MMD (gaussian RBF) loss, MI455X / gfx1250.
//   mean(Kxx) + mean(Kyy) - 2*mean(Kxy),  K = exp(-0.5 * ||a-b||^2)
// Gram matrices via bf16 hi/lo-split WMMA (v_wmma_f32_16x16x32_bf16, f32 acc),
// triangular tiling for the symmetric xx / yy terms, double-buffered LDS with
// global prefetch so L2 latency + conversion VALU hide under the XDL shadow.
// ---------------------------------------------------------------------------

typedef __bf16 bf16_t;
typedef __attribute__((ext_vector_type(16))) __bf16 v16bf;
typedef __attribute__((ext_vector_type(8)))  __bf16 v8bf;
typedef __attribute__((ext_vector_type(4)))  __bf16 v4bf;
typedef __attribute__((ext_vector_type(8)))  float  v8f;

#define NROWS 8192
#define DIMK  512
// -(1/(2*sigma^2)) * log2(e), sigma = 1  ->  exp(-0.5*d2) = exp2(C*d2)
#define NEG_BETA_LOG2E (-0.7213475204444817f)

#define BM 128
#define BN 128
#define BK 32
#define NK (DIMK / BK)        // 16 K-steps
#define ASTRIDE 40            // LDS row stride (elems) for A tiles, layout [m][k]
#define BSTRIDE 136           // LDS row stride (elems) for B tiles, layout [k][n]
#define NTILES (NROWS / BM)   // 64

// ---------------------------------------------------------------------------
// Row squared-norms: one wave32 per row.
// ---------------------------------------------------------------------------
__global__ __launch_bounds__(256) void row_sqnorm(const float* __restrict__ X,
                                                  float* __restrict__ out) {
  const int row  = blockIdx.x * 8 + (threadIdx.x >> 5);
  const int lane = threadIdx.x & 31;
  const float4* p = (const float4*)(X + (size_t)row * DIMK);
  float s = 0.0f;
#pragma unroll
  for (int i = 0; i < 4; ++i) {
    float4 v = p[lane + i * 32];
    s += v.x * v.x + v.y * v.y + v.z * v.z + v.w * v.w;
  }
#pragma unroll
  for (int off = 16; off > 0; off >>= 1) s += __shfl_down(s, off, 32);
  if (lane == 0) out[row] = s;
}

// -------------------------- staging helpers --------------------------------
struct G8 { float4 a[4]; float4 b[4]; };

__device__ __forceinline__ G8 load_g(const float* __restrict__ A,
                                     const float* __restrict__ B,
                                     int rowBase, int colBase, int k0, int t) {
  G8 g;
#pragma unroll
  for (int it = 0; it < 4; ++it) {
    const int f  = t + it * 256;        // 0..1023 over 128 rows x 8 float4s
    const int m  = f >> 3;
    const int c4 = (f & 7) << 2;
    g.a[it] = *(const float4*)(A + (size_t)(rowBase + m) * DIMK + k0 + c4);
    g.b[it] = *(const float4*)(B + (size_t)(colBase + m) * DIMK + k0 + c4);
  }
  return g;
}

__device__ __forceinline__ void store_g(const G8& g, int t,
                                        bf16_t* __restrict__ aHi, bf16_t* __restrict__ aLo,
                                        bf16_t* __restrict__ bHi, bf16_t* __restrict__ bLo) {
#pragma unroll
  for (int it = 0; it < 4; ++it) {
    const int f  = t + it * 256;
    const int m  = f >> 3;
    const int c4 = (f & 7) << 2;
    // A: [m][k] layout, vectorized b64 stores
    {
      const float xs[4] = {g.a[it].x, g.a[it].y, g.a[it].z, g.a[it].w};
      bf16_t h[4], l[4];
#pragma unroll
      for (int e = 0; e < 4; ++e) {
        h[e] = (bf16_t)xs[e];
        l[e] = (bf16_t)(xs[e] - (float)h[e]);
      }
      v4bf h4 = {h[0], h[1], h[2], h[3]};
      v4bf l4 = {l[0], l[1], l[2], l[3]};
      *(v4bf*)(aHi + m * ASTRIDE + c4) = h4;
      *(v4bf*)(aLo + m * ASTRIDE + c4) = l4;
    }
    // B: transposed [k][n] layout (WMMA B operand wants K across lanes)
    {
      const float xs[4] = {g.b[it].x, g.b[it].y, g.b[it].z, g.b[it].w};
#pragma unroll
      for (int e = 0; e < 4; ++e) {
        bf16_t h = (bf16_t)xs[e];
        bf16_t l = (bf16_t)(xs[e] - (float)h);
        bHi[(c4 + e) * BSTRIDE + m] = h;
        bLo[(c4 + e) * BSTRIDE + m] = l;
      }
    }
  }
}

// -------------------------- WMMA compute step ------------------------------
__device__ __forceinline__ void compute_step(
    const bf16_t* __restrict__ aHi, const bf16_t* __restrict__ aLo,
    const bf16_t* __restrict__ bHi, const bf16_t* __restrict__ bLo,
    int waveRow, int waveCol, int lane, int lrow, int kb, v8f acc[2][4]) {
  // A frag (16x32 bf16): lane<16 -> row M=lane, K {0..7,16..23};
  //                      lane>=16 -> same M, K {8..15,24..31}.
  v16bf ah[2], al[2], bh[4], bl[4];
#pragma unroll
  for (int i = 0; i < 2; ++i) {
    const int m = waveRow * 32 + i * 16 + lrow;
    const bf16_t* ph = aHi + m * ASTRIDE + kb;
    const bf16_t* pl = aLo + m * ASTRIDE + kb;
    v8bf a0 = *(const v8bf*)ph, a1 = *(const v8bf*)(ph + 16);
    ah[i] = __builtin_shufflevector(a0, a1, 0,1,2,3,4,5,6,7,8,9,10,11,12,13,14,15);
    v8bf c0 = *(const v8bf*)pl, c1 = *(const v8bf*)(pl + 16);
    al[i] = __builtin_shufflevector(c0, c1, 0,1,2,3,4,5,6,7,8,9,10,11,12,13,14,15);
  }
  // B frag (32x16 bf16): lane L holds column N=L%16, K {kb..kb+7, kb+16..kb+23}
  // (same K order as A), gathered from sB[k][n].
#pragma unroll
  for (int j = 0; j < 4; ++j) {
    const int n0 = waveCol * 64 + j * 16;
    const bf16_t* ph = bHi + kb * BSTRIDE + n0 + lrow;
    const bf16_t* pl = bLo + kb * BSTRIDE + n0 + lrow;
#pragma unroll
    for (int s = 0; s < 16; ++s) {
      const int kk = (s < 8) ? s : (s + 8);
      bh[j][s] = ph[kk * BSTRIDE];
      bl[j][s] = pl[kk * BSTRIDE];
    }
  }
  // 3-term split MACs: cross terms first, hi*hi last
#pragma unroll
  for (int i = 0; i < 2; ++i)
#pragma unroll
    for (int j = 0; j < 4; ++j)
      acc[i][j] = __builtin_amdgcn_wmma_f32_16x16x32_bf16(
          false, ah[i], false, bl[j], (short)0, acc[i][j], false, false);
#pragma unroll
  for (int i = 0; i < 2; ++i)
#pragma unroll
    for (int j = 0; j < 4; ++j)
      acc[i][j] = __builtin_amdgcn_wmma_f32_16x16x32_bf16(
          false, al[i], false, bh[j], (short)0, acc[i][j], false, false);
#pragma unroll
  for (int i = 0; i < 2; ++i)
#pragma unroll
    for (int j = 0; j < 4; ++j)
      acc[i][j] = __builtin_amdgcn_wmma_f32_16x16x32_bf16(
          false, ah[i], false, bh[j], (short)0, acc[i][j], false, false);
}

// ---------------------------------------------------------------------------
// One 128x128 tile of sum(exp(-beta*max(a2+b2-2*A.B^T, 0))).
// 256 threads = 8 waves; wave grid 4(rows of 32) x 2(cols of 64);
// each wave owns 2x4 = 8 v8f accumulators (16x16 WMMA tiles).
// Double-buffered LDS, one barrier per K-step, global prefetch before WMMA.
// triangular!=0: skip tiles below diagonal, weight off-diagonal tiles by 2.
// ---------------------------------------------------------------------------
__global__ __launch_bounds__(256) void mmd_tile_kernel(
    const float* __restrict__ A, const float* __restrict__ B,
    const float* __restrict__ a2, const float* __restrict__ b2,
    float* __restrict__ partials, int triangular) {
  __shared__ __attribute__((aligned(16))) bf16_t sAhi[2][BM * ASTRIDE];
  __shared__ __attribute__((aligned(16))) bf16_t sAlo[2][BM * ASTRIDE];
  __shared__ __attribute__((aligned(16))) bf16_t sBhi[2][BK * BSTRIDE];
  __shared__ __attribute__((aligned(16))) bf16_t sBlo[2][BK * BSTRIDE];
  __shared__ float sa2[BM];
  __shared__ float sb2[BN];
  __shared__ float red[256];

  const int bc = blockIdx.x;   // column tile
  const int br = blockIdx.y;   // row tile
  const int t  = threadIdx.x;
  const int pidx = br * NTILES + bc;

  float weight = 1.0f;
  if (triangular) {
    if (bc < br) {             // below diagonal: covered by transpose tile
      if (t == 0) partials[pidx] = 0.0f;
      return;
    }
    if (bc > br) weight = 2.0f;
  }

  const int rowBase = br * BM;
  const int colBase = bc * BN;

  if (t < BM) sa2[t] = a2[rowBase + t];
  else        sb2[t - BM] = b2[colBase + (t - BM)];

  const int wave    = t >> 5;
  const int lane    = t & 31;
  const int waveRow = wave & 3;          // 0..3 -> 32-row groups
  const int waveCol = wave >> 2;         // 0..1 -> 64-col groups
  const int lrow    = lane & 15;
  const int kb      = (lane >> 4) << 3;  // A-frag K base: 0 or 8

  const v8f zero8 = {0.f, 0.f, 0.f, 0.f, 0.f, 0.f, 0.f, 0.f};
  v8f acc[2][4];
#pragma unroll
  for (int i = 0; i < 2; ++i)
#pragma unroll
    for (int j = 0; j < 4; ++j) acc[i][j] = zero8;

  // prologue: stage K-step 0 into buffer 0
  {
    G8 g = load_g(A, B, rowBase, colBase, 0, t);
    store_g(g, t, sAhi[0], sAlo[0], sBhi[0], sBlo[0]);
  }
  __syncthreads();

  // steady state: prefetch k+1, WMMA on k, convert+store k+1, one barrier
#pragma unroll 2
  for (int ks = 0; ks < NK - 1; ++ks) {
    const int cur = ks & 1, nxt = cur ^ 1;
    G8 g = load_g(A, B, rowBase, colBase, (ks + 1) * BK, t);   // issues early
    compute_step(sAhi[cur], sAlo[cur], sBhi[cur], sBlo[cur],
                 waveRow, waveCol, lane, lrow, kb, acc);
    store_g(g, t, sAhi[nxt], sAlo[nxt], sBhi[nxt], sBlo[nxt]);
    __syncthreads();
  }
  // last K-step
  compute_step(sAhi[(NK - 1) & 1], sAlo[(NK - 1) & 1],
               sBhi[(NK - 1) & 1], sBlo[(NK - 1) & 1],
               waveRow, waveCol, lane, lrow, kb, acc);

  // ---- epilogue: kernel values + block reduction --------------------------
  // C layout: VGPR v, lane<16 -> (M=v, N=lane); lane>=16 -> (M=v+8, N=lane-16)
  const int half = lane >> 4;
  float lsum = 0.0f;
#pragma unroll
  for (int i = 0; i < 2; ++i) {
    const int mBase = waveRow * 32 + i * 16 + half * 8;
#pragma unroll
    for (int j = 0; j < 4; ++j) {
      const int n = waveCol * 64 + j * 16 + lrow;
      const float bn = sb2[n];
#pragma unroll
      for (int v = 0; v < 8; ++v) {
        float d2 = sa2[mBase + v] + bn - 2.0f * acc[i][j][v];
        d2 = fmaxf(d2, 0.0f);
        lsum += __builtin_amdgcn_exp2f(NEG_BETA_LOG2E * d2);
      }
    }
  }

  red[t] = lsum;
  __syncthreads();
#pragma unroll
  for (int s = 128; s > 0; s >>= 1) {
    if (t < s) red[t] += red[t + s];
    __syncthreads();
  }
  if (t == 0) partials[pidx] = weight * red[0];
}

// ---------------------------------------------------------------------------
// Deterministic final combine: out = (Sxx + Syy - 2*Sxy) / N^2
// ---------------------------------------------------------------------------
__global__ __launch_bounds__(256) void mmd_reduce(const float* __restrict__ part,
                                                  float* __restrict__ out) {
  __shared__ float red[256];
  const int t   = threadIdx.x;
  const int NT2 = NTILES * NTILES;  // 4096
  float v = 0.0f;
  for (int i = t; i < NT2; i += 256)
    v += part[i] + part[NT2 + i] - 2.0f * part[2 * NT2 + i];
  red[t] = v;
  __syncthreads();
#pragma unroll
  for (int s = 128; s > 0; s >>= 1) {
    if (t < s) red[t] += red[t + s];
    __syncthreads();
  }
  if (t == 0) out[0] = red[0] * (1.0f / ((float)NROWS * (float)NROWS));
}

// ---------------------------------------------------------------------------
extern "C" void kernel_launch(void* const* d_in, const int* in_sizes, int n_in,
                              void* d_out, int out_size, void* d_ws, size_t ws_size,
                              hipStream_t stream) {
  (void)in_sizes; (void)n_in; (void)out_size; (void)ws_size;
  const float* x = (const float*)d_in[0];
  const float* y = (const float*)d_in[1];
  float* ws   = (float*)d_ws;
  float* xs2  = ws;                       // 8192 floats
  float* ys2  = ws + NROWS;               // 8192 floats
  float* part = ws + 2 * NROWS;           // 3 * 4096 floats
  float* out  = (float*)d_out;

  row_sqnorm<<<dim3(NROWS / 8), 256, 0, stream>>>(x, xs2);
  row_sqnorm<<<dim3(NROWS / 8), 256, 0, stream>>>(y, ys2);

  dim3 grid(NTILES, NTILES);
  mmd_tile_kernel<<<grid, 256, 0, stream>>>(x, x, xs2, xs2, part, 1);
  mmd_tile_kernel<<<grid, 256, 0, stream>>>(y, y, ys2, ys2, part + NTILES * NTILES, 1);
  mmd_tile_kernel<<<grid, 256, 0, stream>>>(x, y, xs2, ys2, part + 2 * NTILES * NTILES, 0);

  mmd_reduce<<<1, 256, 0, stream>>>(part, out);
}
